// Encoder_53678501266015
// MI455X (gfx1250) — compile-verified, matrix operand order checked by reading, not run on hardware
//
#include <hip/hip_runtime.h>
#include <hip/hip_bf16.h>
#include <stdint.h>

// ---------------------------------------------------------------------------
// CDNA5 (gfx1250) implicit-GEMM masked 3D convolution, wave32 WMMA.
//
//   - activations: f16 channel-last [vox][Cpad], Cpad always a multiple of 32
//                  -> A fragment is exactly 2x aligned global_load_b128 / lane
//   - weights:     pre-swizzled f16 B fragments, staged in LDS once per block
//                  (8 waves share them) -> inner loop reads via ds_load_b128
//   - loop:        stencil-position outer (uniform SALU), 32-ch K-blocks inner
//   - tiling:      32 voxels x 16 cout per wave, 2 accumulators share B
//   - bounds:      only k=3 SAME can go OOB -> templated out elsewhere
//   - epilogue:    fused bias + occupancy-mask + ReLU; pad channels get zeros
// ---------------------------------------------------------------------------

typedef __attribute__((ext_vector_type(16))) _Float16 v16h;
typedef __attribute__((ext_vector_type(8)))  _Float16 v8h;
typedef __attribute__((ext_vector_type(8)))  float    v8f;

template <int K, bool BCHECK>
__global__ __launch_bounds__(256)
void conv3d_wmma(const _Float16* __restrict__ in,    // [D^3][Cin] f16, Cin % 32 == 0
                 const _Float16* __restrict__ wswz,  // [nTile][kvol][Cin/32][lane][16]
                 const float* __restrict__ bias,     // [Cout] (real)
                 const float* __restrict__ mask,     // [OD^3] 0/1 f32
                 _Float16* __restrict__ outH,        // [OD^3][CoutTot]
                 float* __restrict__ outF,           // [Cout][OD^3] (final layer)
                 int Cin, int Cout, int coutBase, int CoutTot,
                 int D, int OD, int stride, int doRelu, int writeF32)
{
    extern __shared__ _Float16 smem[];               // kvol * (Cin/32) * 512 halves

    const int pad  = BCHECK ? 1 : 0;     // k3 SAME; k2/s2 and k1 are always in range
    const int lane = threadIdx.x & 31;
    const int wave = threadIdx.x >> 5;
    const int lid  = lane & 15;
    const int hi   = lane >> 4;
    const int kvol = K * K * K;
    const int nCb  = Cin >> 5;

    // --- Stage this Cout-tile's B fragments into LDS (shared by all 8 waves) ---
    {
        const v16h* src = (const v16h*)(wswz + (size_t)blockIdx.y * kvol * nCb * 512);
        v16h* dst = (v16h*)smem;
        const int totalV16 = kvol * nCb * 32;
        for (int t = threadIdx.x; t < totalV16; t += blockDim.x)
            dst[t] = src[t];
    }
    __syncthreads();

    const int mBase = (blockIdx.x * (blockDim.x >> 5) + wave) * 32;  // nvox % 256 == 0
    const int ncol  = blockIdx.y * 16 + lid;

    // Decode the two A-row voxels owned by this lane (once).
    const int vm0 = mBase + lid, vm1 = mBase + 16 + lid;
    const int od0 = vm0 / (OD * OD), r0 = vm0 - od0 * OD * OD, oh0 = r0 / OD, ow0 = r0 - oh0 * OD;
    const int od1 = vm1 / (OD * OD), r1 = vm1 - od1 * OD * OD, oh1 = r1 / OD, ow1 = r1 - oh1 * OD;
    const int bz0 = od0 * stride - pad, by0 = oh0 * stride - pad, bx0 = ow0 * stride - pad;
    const int bz1 = od1 * stride - pad, by1 = oh1 * stride - pad, bx1 = ow1 * stride - pad;

    const v16h* bfrag = (const v16h*)smem + lane;    // advance by 32 per fragment
    const int hi8 = hi * 8;

    v8f acc0 = {}, acc1 = {};
    int kz = 0, ky = 0, kx = 0;
    for (int s = 0; s < kvol; ++s) {
        const int iz0 = bz0 + kz, iy0 = by0 + ky, ix0 = bx0 + kx;
        const int iz1 = bz1 + kz, iy1 = by1 + ky, ix1 = bx1 + kx;
        bool inb0 = true, inb1 = true;
        int soff0, soff1;
        if (BCHECK) {
            inb0 = ((unsigned)iz0 < (unsigned)D) & ((unsigned)iy0 < (unsigned)D) &
                   ((unsigned)ix0 < (unsigned)D);
            inb1 = ((unsigned)iz1 < (unsigned)D) & ((unsigned)iy1 < (unsigned)D) &
                   ((unsigned)ix1 < (unsigned)D);
            soff0 = inb0 ? ((iz0 * D + iy0) * D + ix0) : 0;
            soff1 = inb1 ? ((iz1 * D + iy1) * D + ix1) : 0;
        } else {
            soff0 = (iz0 * D + iy0) * D + ix0;
            soff1 = (iz1 * D + iy1) * D + ix1;
        }
        const _Float16* a0p = in + (size_t)soff0 * Cin + hi8;
        const _Float16* a1p = in + (size_t)soff1 * Cin + hi8;
        // Speculative next-line prefetch: next stencil x-offset is +Cin halves.
        __builtin_prefetch(a0p + Cin, 0, 0);
        __builtin_prefetch(a1p + Cin, 0, 0);

        for (int cb = 0; cb < Cin; cb += 32) {
            v16h B = *bfrag;                              // ds_load_b128 x2
            bfrag += 32;
            v8h x00 = *(const v8h*)(a0p + cb);            // channels cb+hi8 .. +7
            v8h x01 = *(const v8h*)(a0p + cb + 16);       // channels cb+hi8+16 .. +23
            v8h x10 = *(const v8h*)(a1p + cb);
            v8h x11 = *(const v8h*)(a1p + cb + 16);
            v16h A0, A1;
#pragma unroll
            for (int j = 0; j < 8; ++j) {
                A0[j] = x00[j]; A0[8 + j] = x01[j];
                A1[j] = x10[j]; A1[8 + j] = x11[j];
            }
            if (BCHECK) {
#pragma unroll
                for (int j = 0; j < 16; ++j) {
                    A0[j] = inb0 ? A0[j] : (_Float16)0.f;   // v_cndmask
                    A1[j] = inb1 ? A1[j] : (_Float16)0.f;
                }
            }
            acc0 = __builtin_amdgcn_wmma_f32_16x16x32_f16(false, A0, false, B,
                                                          (short)0, acc0, false, false);
            acc1 = __builtin_amdgcn_wmma_f32_16x16x32_f16(false, A1, false, B,
                                                          (short)0, acc1, false, false);
        }
        if (++kx == K) { kx = 0; if (++ky == K) { ky = 0; ++kz; } }   // uniform SALU
    }

    const int nvoxOut = OD * OD * OD;
    const float bval = (ncol < Cout) ? bias[ncol] : 0.f;   // pad channels -> 0
#pragma unroll
    for (int r = 0; r < 8; ++r) {
        const int m  = r + hi * 8;                 // C/D layout: M = vgpr + 8*hi, N = lid
        const int v0 = mBase + m, v1 = mBase + 16 + m;
        float valA = (acc0[r] + bval) * mask[v0];
        float valB = (acc1[r] + bval) * mask[v1];
        if (doRelu) { valA = fmaxf(valA, 0.f); valB = fmaxf(valB, 0.f); }
        if (writeF32) {
            outF[(size_t)(coutBase + ncol) * nvoxOut + v0] = valA;
            outF[(size_t)(coutBase + ncol) * nvoxOut + v1] = valB;
        } else {
            outH[(size_t)v0 * CoutTot + coutBase + ncol] = (_Float16)valA;
            outH[(size_t)v1 * CoutTot + coutBase + ncol] = (_Float16)valB;
        }
    }
}

// ---------------------------------------------------------------------------
// Cin==1 first layer: K-total = 27 fits in ONE WMMA (K index = stencil pos).
// ---------------------------------------------------------------------------
__global__ __launch_bounds__(256)
void conv3d_cin1(const _Float16* __restrict__ xh,    // [D^3] single channel f16
                 const _Float16* __restrict__ wswz,  // [nTile][lane][16]
                 const float* __restrict__ bias,
                 const float* __restrict__ mask,
                 _Float16* __restrict__ outH,        // [D^3][CoutTot]
                 int Cout, int CoutTot, int D, int doRelu)
{
    const int lane = threadIdx.x & 31;
    const int wave = threadIdx.x >> 5;
    const int lid  = lane & 15;
    const int hi   = lane >> 4;
    const int mBase = (blockIdx.x * (blockDim.x >> 5) + wave) * 32;
    const int ncol  = blockIdx.y * 16 + lid;

    const int vm0 = mBase + lid, vm1 = mBase + 16 + lid;
    const int od0 = vm0 / (D * D), r0 = vm0 - od0 * D * D, oh0 = r0 / D, ow0 = r0 - oh0 * D;
    const int od1 = vm1 / (D * D), r1 = vm1 - od1 * D * D, oh1 = r1 / D, ow1 = r1 - oh1 * D;

    v16h A0, A1;
#pragma unroll
    for (int i = 0; i < 16; ++i) {
        const int sv = ((i < 8) ? i : (i + 8)) + hi * 8;   // K slot -> stencil position
        _Float16 a0 = (_Float16)0.f, a1 = (_Float16)0.f;
        if (sv < 27) {
            const int kz = sv / 9, rr = sv - kz * 9, ky = rr / 3, kx = rr - ky * 3;
            const int iz0 = od0 + kz - 1, iy0 = oh0 + ky - 1, ix0 = ow0 + kx - 1;
            const int iz1 = od1 + kz - 1, iy1 = oh1 + ky - 1, ix1 = ow1 + kx - 1;
            if (((unsigned)iz0 < (unsigned)D) & ((unsigned)iy0 < (unsigned)D) &
                ((unsigned)ix0 < (unsigned)D))
                a0 = xh[((size_t)iz0 * D + iy0) * D + ix0];
            if (((unsigned)iz1 < (unsigned)D) & ((unsigned)iy1 < (unsigned)D) &
                ((unsigned)ix1 < (unsigned)D))
                a1 = xh[((size_t)iz1 * D + iy1) * D + ix1];
        }
        A0[i] = a0; A1[i] = a1;
    }
    v16h B = ((const v16h*)wswz)[(size_t)blockIdx.y * 32 + lane];
    v8f acc0 = {}, acc1 = {};
    acc0 = __builtin_amdgcn_wmma_f32_16x16x32_f16(false, A0, false, B, (short)0, acc0, false, false);
    acc1 = __builtin_amdgcn_wmma_f32_16x16x32_f16(false, A1, false, B, (short)0, acc1, false, false);

    const float bval = (ncol < Cout) ? bias[ncol] : 0.f;
#pragma unroll
    for (int r = 0; r < 8; ++r) {
        const int m = r + hi * 8;
        const int v0 = mBase + m, v1 = mBase + 16 + m;
        float valA = (acc0[r] + bval) * mask[v0];
        float valB = (acc1[r] + bval) * mask[v1];
        if (doRelu) { valA = fmaxf(valA, 0.f); valB = fmaxf(valB, 0.f); }
        outH[(size_t)v0 * CoutTot + ncol] = (_Float16)valA;
        outH[(size_t)v1 * CoutTot + ncol] = (_Float16)valB;
    }
}

// ---------------------------------------------------------------------------
// Weight swizzles
// ---------------------------------------------------------------------------
__global__ void repack_w(const float* __restrict__ w, _Float16* __restrict__ o,
                         int Cin, int Cout, int kvol, int nTiles)
{
    const int nCb = Cin >> 5;
    const int total = nTiles * kvol * nCb * 512;
    int i = blockIdx.x * blockDim.x + threadIdx.x;
    if (i >= total) return;
    int slot = i & 15;
    int lane = (i >> 4) & 31;
    int frag = i >> 9;
    int cbI  = frag % nCb;
    int rst  = frag / nCb;
    int s    = rst % kvol;
    int tile = rst / kvol;
    int lid = lane & 15, hi = lane >> 4;
    int n = tile * 16 + lid;
    int c = cbI * 32 + ((slot < 8) ? slot : slot + 8) + hi * 8;
    float v = 0.f;
    if (n < Cout && c < Cin) v = w[((size_t)n * Cin + c) * kvol + s];
    o[i] = (_Float16)v;
}

__global__ void repack_w1(const float* __restrict__ w, _Float16* __restrict__ o,
                          int Cout, int kvol, int nTiles)   // Cin == 1
{
    const int total = nTiles * 512;
    int i = blockIdx.x * blockDim.x + threadIdx.x;
    if (i >= total) return;
    int slot = i & 15;
    int lane = (i >> 4) & 31;
    int tile = i >> 9;
    int lid = lane & 15, hi = lane >> 4;
    int n = tile * 16 + lid;
    int s = ((slot < 8) ? slot : slot + 8) + hi * 8;
    float v = 0.f;
    if (n < Cout && s < kvol) v = w[(size_t)n * kvol + s];
    o[i] = (_Float16)v;
}

// ---------------------------------------------------------------------------
// Elementwise helpers
// ---------------------------------------------------------------------------
__global__ void cvt_f32_to_f16(const float* __restrict__ x, _Float16* __restrict__ o, int n)
{
    int i = blockIdx.x * blockDim.x + threadIdx.x;
    if (i < n) o[i] = (_Float16)x[i];
}

__global__ void mask_to_f32(const uint8_t* __restrict__ m, float* __restrict__ out, int n)
{
    int i = blockIdx.x * blockDim.x + threadIdx.x;
    if (i < n) out[i] = m[i] ? 1.f : 0.f;
}

__global__ void mask_down(const float* __restrict__ in, float* __restrict__ out, int OD)
{
    int n = OD * OD * OD;
    int i = blockIdx.x * blockDim.x + threadIdx.x;
    if (i >= n) return;
    int od = i / (OD * OD), r = i - od * OD * OD, oh = r / OD, ow = r - oh * OD;
    int Din = OD * 2;
    float v = 0.f;
#pragma unroll
    for (int dz = 0; dz < 2; ++dz)
#pragma unroll
        for (int dy = 0; dy < 2; ++dy)
#pragma unroll
            for (int dx = 0; dx < 2; ++dx)
                v = fmaxf(v, in[((size_t)(od * 2 + dz) * Din + (oh * 2 + dy)) * Din + (ow * 2 + dx)]);
    out[i] = v;
}

// Vectorized f16 residual add (n % 8 == 0) -> v_pk_add_f16.
__global__ void add_inplace_h8(_Float16* __restrict__ dst, const _Float16* __restrict__ src, int n8)
{
    int i = blockIdx.x * blockDim.x + threadIdx.x;
    if (i < n8) {
        v8h d = ((v8h*)dst)[i];
        v8h s = ((const v8h*)src)[i];
        d += s;
        ((v8h*)dst)[i] = d;
    }
}

// ---------------------------------------------------------------------------
// Host-side sequencing
// ---------------------------------------------------------------------------
struct CP { const float* w; const float* b; };

extern "C" void kernel_launch(void* const* d_in, const int* in_sizes, int n_in,
                              void* d_out, int out_size, void* d_ws, size_t ws_size,
                              hipStream_t stream)
{
    (void)in_sizes; (void)n_in; (void)out_size; (void)ws_size;

    const float*   x     = (const float*)d_in[0];
    const uint8_t* mask0 = (const uint8_t*)d_in[1];

    int idx = 2;
    auto take = [&](void) -> CP {
        CP c{ (const float*)d_in[idx], (const float*)d_in[idx + 1] };
        idx += 2;
        return c;
    };
    CP b0_c0 = take(), b0_c1 = take();
    CP b0_irb[3][5];
    for (int i = 0; i < 3; ++i) for (int j = 0; j < 5; ++j) b0_irb[i][j] = take();
    CP b1_c0 = take(), b1_c1 = take();
    CP b1_irb[3][5];
    for (int i = 0; i < 3; ++i) for (int j = 0; j < 5; ++j) b1_irb[i][j] = take();
    CP b2_c0 = take(), b2_c1 = take();
    CP b2_irb[3][5];                  // b2_irb precedes b2_c2 in the params dict
    for (int i = 0; i < 3; ++i) for (int j = 0; j < 5; ++j) b2_irb[i][j] = take();
    CP b2_c2 = take();

    uint8_t* ws = (uint8_t*)d_ws;
    size_t off = 0;
    auto alloc = [&](size_t bytes) -> void* {
        void* p = (void*)(ws + off);
        off = (off + bytes + 255) & ~(size_t)255;
        return p;
    };
    const int V128 = 128 * 128 * 128, V64 = 64 * 64 * 64, V32 = 32 * 32 * 32, V16 = 16 * 16 * 16;
    float* m0f = (float*)alloc((size_t)V128 * 4);
    float* m1  = (float*)alloc((size_t)V64 * 4);
    float* m2  = (float*)alloc((size_t)V32 * 4);
    float* m3  = (float*)alloc((size_t)V16 * 4);
    _Float16* xh = (_Float16*)alloc((size_t)V128 * 2);          // x as f16 (1 channel)
    _Float16* A0 = (_Float16*)alloc((size_t)V128 * 32 * 2);     // 16ch padded to 32
    _Float16* B1 = (_Float16*)alloc((size_t)V64 * 32 * 2);      // block0 ping
    _Float16* B2 = (_Float16*)alloc((size_t)V64 * 32 * 2);      // block0 pong
    _Float16* T1 = (_Float16*)alloc((size_t)V64 * 32 * 2);      // IRB temp (padded 32)
    _Float16* T2 = (_Float16*)alloc((size_t)V64 * 32 * 2);      // IRB temp (padded 32)
    _Float16* wswz = (_Float16*)alloc((size_t)256 * 1024);      // swizzled weights
    _Float16* C1 = A0;                                          // block1 ping/pong in A0
    _Float16* C2 = A0 + (size_t)V32 * 64;

    // Generic conv: Cin is the PADDED channel count of the input buffer (mult of 32).
    auto conv = [&](const _Float16* in, CP p, const float* mk,
                    _Float16* outH, float* outF,
                    int Cin, int Cout, int CoutPad, int coutBase, int CoutTot,
                    int D, int OD, int k, int st, int relu) {
        int kvol = k * k * k;
        int nTiles = CoutPad / 16;
        int nCb = Cin / 32;
        int total = nTiles * kvol * nCb * 512;
        repack_w<<<(total + 255) / 256, 256, 0, stream>>>(p.w, wswz, Cin, Cout, kvol, nTiles);
        int nvox = OD * OD * OD;
        dim3 grid(nvox / 256, nTiles);
        size_t shmem = (size_t)kvol * nCb * 512 * sizeof(_Float16);  // B fragments
        int wf = outF != nullptr;
        if (k == 3)
            conv3d_wmma<3, true><<<grid, 256, shmem, stream>>>(in, wswz, p.b, mk, outH, outF,
                Cin, Cout, coutBase, CoutTot, D, OD, st, relu, wf);
        else if (k == 2)
            conv3d_wmma<2, false><<<grid, 256, shmem, stream>>>(in, wswz, p.b, mk, outH, outF,
                Cin, Cout, coutBase, CoutTot, D, OD, st, relu, wf);
        else
            conv3d_wmma<1, false><<<grid, 256, shmem, stream>>>(in, wswz, p.b, mk, outH, outF,
                Cin, Cout, coutBase, CoutTot, D, OD, st, relu, wf);
    };

    // IRB: temps stored padded to 32 channels; consumed with Cin=32.
    auto irb = [&](CP (&p)[5], int C, int dim, const float* mk, _Float16*& cur, _Float16*& alt) {
        int nv = dim * dim * dim;
        conv(cur, p[0], mk, T1, nullptr, C,  C / 4, 32,    0,     32, dim, dim, 3, 1, 1);
        conv(T1,  p[1], mk, alt, nullptr, 32, C / 2, C / 2, 0,     C,  dim, dim, 3, 1, 0);
        conv(cur, p[2], mk, T1, nullptr, C,  C / 4, 32,    0,     32, dim, dim, 1, 1, 1);
        conv(T1,  p[3], mk, T2, nullptr, 32, C / 4, 32,    0,     32, dim, dim, 3, 1, 1);
        conv(T2,  p[4], mk, alt, nullptr, 32, C / 2, C / 2, C / 2, C,  dim, dim, 1, 1, 0);
        int n8 = C * nv / 8;
        add_inplace_h8<<<(n8 + 255) / 256, 256, 0, stream>>>(alt, cur, n8);
        _Float16* t = cur; cur = alt; alt = t;
    };

    // ---- masks + input conversion ----
    mask_to_f32<<<(V128 + 255) / 256, 256, 0, stream>>>(mask0, m0f, V128);
    mask_down<<<(V64 + 255) / 256, 256, 0, stream>>>(m0f, m1, 64);
    mask_down<<<(V32 + 255) / 256, 256, 0, stream>>>(m1, m2, 32);
    mask_down<<<(V16 + 255) / 256, 256, 0, stream>>>(m2, m3, 16);
    cvt_f32_to_f16<<<(V128 + 255) / 256, 256, 0, stream>>>(x, xh, V128);

    // ---- block0 ----  (Cin=1 special kernel; output padded to 32 channels)
    repack_w1<<<(2 * 512 + 255) / 256, 256, 0, stream>>>(b0_c0.w, wswz, 16, 27, 2);
    {
        dim3 grid(V128 / 256, 2);
        conv3d_cin1<<<grid, 256, 0, stream>>>(xh, wswz, b0_c0.b, m0f, A0, 16, 32, 128, 1);
    }
    conv(A0, b0_c1, m1, B1, nullptr, 32, 32, 32, 0, 32, 128, 64, 2, 2, 1);
    _Float16* cur = B1; _Float16* alt = B2;
    for (int i = 0; i < 3; ++i) irb(b0_irb[i], 32, 64, m1, cur, alt);

    // ---- block1 ----
    conv(cur, b1_c0, m1, alt, nullptr, 32, 32, 32, 0, 32, 64, 64, 3, 1, 1);
    conv(alt, b1_c1, m2, C1, nullptr, 32, 64, 64, 0, 64, 64, 32, 2, 2, 1);
    _Float16* cur2 = C1; _Float16* alt2 = C2;
    for (int i = 0; i < 3; ++i) irb(b1_irb[i], 64, 32, m2, cur2, alt2);

    // ---- block2 ----
    conv(cur2, b2_c0, m2, alt2, nullptr, 64, 64, 64, 0, 64, 32, 32, 3, 1, 1);
    conv(alt2, b2_c1, m3, B1, nullptr, 64, 32, 32, 0, 32, 32, 16, 2, 2, 1);
    _Float16* cur3 = B1; _Float16* alt3 = B2;
    for (int i = 0; i < 3; ++i) irb(b2_irb[i], 32, 16, m3, cur3, alt3);
    conv(cur3, b2_c2, m3, nullptr, (float*)d_out, 32, 32, 32, 0, 32, 16, 16, 3, 1, 0);
}
